// MNIST_SNN_BURST_74182675136761
// MI455X (gfx1250) — matrix operand order repaired and checked by reading, not compiled, not run on hardware
//
#include <hip/hip_runtime.h>

typedef __attribute__((ext_vector_type(16))) _Float16 v16h;
typedef __attribute__((ext_vector_type(8)))  _Float16 v8h;
typedef __attribute__((ext_vector_type(8)))  float    v8f;
typedef __attribute__((ext_vector_type(4)))  float    v4f;

#define T_STEPS   100
#define D_IN      784
#define D_H       100
#define D_OUT     10
#define BATCH     1024
#define M_TOTAL   (BATCH * T_STEPS)   /* 102400, divisible by 128 */
#define K1PAD     800                 /* 784 -> 25 chunks of 32 */
#define N1PAD     112                 /* 100 -> 7 tiles of 16 */
#define K2PAD     128                 /* 100 -> 4 chunks of 32 */
#define C1STRIDE  112
#define S1STRIDE  128
#define LDSB_PITCH 40                 /* f16 pitch: conflict-free b128 LDS reads */
#define KSTEPS    25
#define BETA      0.9f

// ---------------- prep: f32 weights -> zero-padded f16 ----------------
__global__ void snn_prep(const float* __restrict__ W1, const float* __restrict__ W2,
                         _Float16* __restrict__ W1h, _Float16* __restrict__ W2h) {
  int i = blockIdx.x * blockDim.x + threadIdx.x;
  const int n1 = N1PAD * K1PAD; // 89600
  if (i < n1) {
    int n = i / K1PAD, k = i - n * K1PAD;
    float v = (n < D_H && k < D_IN) ? W1[n * D_IN + k] : 0.0f;
    W1h[i] = (_Float16)v;
  } else {
    int j = i - n1;
    if (j < 16 * K2PAD) {
      int n = j / K2PAD, k = j - n * K2PAD;
      float v = (n < D_OUT && k < D_H) ? W2[n * D_H + k] : 0.0f;
      W2h[j] = (_Float16)v;
    }
  }
}

// ---------------- GEMM1: CUR1[m][112] = X[m][784] * W1h^T ----------------
// Double-buffered LDS for W1h K-chunks, staged with async global->LDS (ASYNCcnt),
// one barrier per K-step, B fragments register-blocked across 7 N-tiles.
__global__ __launch_bounds__(256) void snn_gemm1(const float* __restrict__ X,
                                                 const _Float16* __restrict__ W1h,
                                                 float* __restrict__ CUR1) {
  __shared__ _Float16 ldsB[2][N1PAD * LDSB_PITCH]; // 2 x 8960 B
  const int tid  = threadIdx.x;
  const int wave = tid >> 5;
  const int lane = tid & 31;
  const int hv   = lane >> 4;   // half-wave select
  const int row  = lane & 15;
  const int m0   = (blockIdx.x * 8 + wave) * 16;

  // per-thread async staging slots: 448 x 16B transfers per chunk, 2 per thread
  const int id2a = tid;                 // always < 448
  const int id2b = tid + 256;           // < 448 only for tid < 192
  const int na = id2a >> 2, qa = id2a & 3;
  const int nb = id2b >> 2, qb = id2b & 3;

  auto stage_async = [&](int kt, int buf) {
    const unsigned kbyte = (unsigned)(kt * 32 * 2);   // K-chunk byte offset in a W1h row
    {
      unsigned lds_a = (unsigned)(size_t)&ldsB[buf][na * LDSB_PITCH] + qa * 16u;
      unsigned voff  = (unsigned)(na * (K1PAD * 2)) + kbyte + qa * 16u;
      asm volatile("global_load_async_to_lds_b128 %0, %1, %2"
                   :: "v"(lds_a), "v"(voff), "s"(W1h) : "memory");
    }
    if (id2b < 448) {
      unsigned lds_b = (unsigned)(size_t)&ldsB[buf][nb * LDSB_PITCH] + qb * 16u;
      unsigned voff  = (unsigned)(nb * (K1PAD * 2)) + kbyte + qb * 16u;
      asm volatile("global_load_async_to_lds_b128 %0, %1, %2"
                   :: "v"(lds_b), "v"(voff), "s"(W1h) : "memory");
    }
  };

  v8f acc[7] = {};
  const float* xrow = X + (size_t)(m0 + row) * D_IN;

  // prologue: stage chunk 0 into buffer 0
  stage_async(0, 0);
  asm volatile("s_wait_asynccnt 0x0" ::: "memory");
  __syncthreads();

  for (int kt = 0; kt < KSTEPS; ++kt) {
    const int k0  = kt * 32;
    const int cur = kt & 1;

    if (kt + 1 < KSTEPS) stage_async(kt + 1, cur ^ 1);   // overlap next-chunk DMA

    // prefetch X chunk three K-steps ahead (speculative; dropped if invalid)
    __builtin_prefetch(xrow + k0 + 96, 0, 0);

    // A fragment: stream X once with non-temporal loads, convert f32->f16 in regs
    const v4f* g1 = (const v4f*)(xrow + k0 + hv * 8);
    v4f f0 = __builtin_nontemporal_load(g1);
    v4f f1 = __builtin_nontemporal_load(g1 + 1);
    v4f f2 = {0.f, 0.f, 0.f, 0.f};
    v4f f3 = {0.f, 0.f, 0.f, 0.f};
    if (k0 + 16 < D_IN) {  // last chunk: upper K half is pure zero-pad
      const v4f* g2 = (const v4f*)(xrow + k0 + 16 + hv * 8);
      f2 = __builtin_nontemporal_load(g2);
      f3 = __builtin_nontemporal_load(g2 + 1);
    }
    v16h a;
    a[0]=(_Float16)f0[0]; a[1]=(_Float16)f0[1]; a[2]=(_Float16)f0[2]; a[3]=(_Float16)f0[3];
    a[4]=(_Float16)f1[0]; a[5]=(_Float16)f1[1]; a[6]=(_Float16)f1[2]; a[7]=(_Float16)f1[3];
    a[8]=(_Float16)f2[0]; a[9]=(_Float16)f2[1]; a[10]=(_Float16)f2[2]; a[11]=(_Float16)f2[3];
    a[12]=(_Float16)f3[0]; a[13]=(_Float16)f3[1]; a[14]=(_Float16)f3[2]; a[15]=(_Float16)f3[3];

    // register-block all 7 B fragments, then back-to-back WMMAs
    v16h bf[7];
#pragma unroll
    for (int nt = 0; nt < 7; ++nt)
      bf[nt] = __builtin_shufflevector(*(const v8h*)&ldsB[cur][(nt * 16 + row) * LDSB_PITCH + hv * 8], *(const v8h*)&ldsB[cur][(nt * 16 + row) * LDSB_PITCH + 16 + hv * 8], 0,1,2,3,4,5,6,7,8,9,10,11,12,13,14,15);
#pragma unroll
    for (int nt = 0; nt < 7; ++nt)
      acc[nt] = __builtin_amdgcn_wmma_f32_16x16x32_f16(
          false, a, false, bf[nt], (short)0, acc[nt], false, false);

    asm volatile("s_wait_asynccnt 0x0" ::: "memory");  // own DMA for kt+1 landed
    __syncthreads();                                    // all waves done reading buf cur
  }

#pragma unroll
  for (int nt = 0; nt < 7; ++nt)
#pragma unroll
    for (int j = 0; j < 8; ++j)
      CUR1[(size_t)(m0 + j + hv * 8) * C1STRIDE + nt * 16 + row] = acc[nt][j];
}

// ---------------- LIF layer 1: recurrence over T, emit f16 spikes ----------------
__global__ void snn_lif1(const float* __restrict__ CUR1, const float* __restrict__ b1,
                         _Float16* __restrict__ SPK1) {
  int id = blockIdx.x * blockDim.x + threadIdx.x;  // B * D_H = 102400
  int b = id / D_H, h = id - b * D_H;
  float bias = b1[h];
  float mem = 0.0f;
  for (int t = 0; t < T_STEPS; ++t) {
    size_t m = (size_t)b * T_STEPS + t;
    float cur = CUR1[m * C1STRIDE + h] + bias;
    mem = BETA * mem + cur;
    float spk = (mem > 1.0f) ? 1.0f : 0.0f;   // spike(mem - 1 > 0)
    mem -= spk;                                // reset by subtraction (thr=1)
    SPK1[m * S1STRIDE + h] = (_Float16)spk;
    if (h < S1STRIDE - D_H) SPK1[m * S1STRIDE + D_H + h] = (_Float16)0.0f; // zero K-pad
  }
}

// ---------------- GEMM2: CUR2[m][16] = SPK1[m][128] * W2h^T ----------------
__global__ __launch_bounds__(256) void snn_gemm2(const _Float16* __restrict__ SPK1,
                                                 const _Float16* __restrict__ W2h,
                                                 float* __restrict__ CUR2) {
  const int tid  = threadIdx.x;
  const int wave = tid >> 5;
  const int lane = tid & 31;
  const int hv   = lane >> 4;
  const int row  = lane & 15;
  const int m0   = (blockIdx.x * 8 + wave) * 16;

  const _Float16* arow = SPK1 + (size_t)(m0 + row) * S1STRIDE;
  const _Float16* brow = W2h + row * K2PAD + hv * 16;
  v8f acc = {};
#pragma unroll
  for (int kt = 0; kt < 4; ++kt) {
    int k0 = kt * 32;
    v8h alo = *(const v8h*)(arow + k0 + hv * 16);
    v8h ahi = *(const v8h*)(arow + k0 + hv * 16 + 8);
    v16h a = __builtin_shufflevector(alo, ahi, 0,1,2,3,4,5,6,7,8,9,10,11,12,13,14,15);
    v16h b = *(const v16h*)(brow + k0);
    acc = __builtin_amdgcn_wmma_f32_16x16x32_f16(
        false, a, false, b, (short)0, acc, false, false);
  }
#pragma unroll
  for (int j = 0; j < 8; ++j)
    CUR2[(size_t)(m0 + j + hv * 8) * 16 + row] = acc[j];
}

// ---------------- LIF layer 2: recurrence over T, write spk_rec [T,B,10] ----------------
__global__ void snn_lif2(const float* __restrict__ CUR2, const float* __restrict__ b2,
                         float* __restrict__ out) {
  int id = blockIdx.x * blockDim.x + threadIdx.x;  // B * D_OUT = 10240
  int b = id / D_OUT, o = id - b * D_OUT;
  float bias = b2[o];
  float mem = 0.0f;
  for (int t = 0; t < T_STEPS; ++t) {
    size_t m = (size_t)b * T_STEPS + t;
    float cur = CUR2[m * 16 + o] + bias;
    mem = BETA * mem + cur;
    float spk = (mem > 1.0f) ? 1.0f : 0.0f;
    mem -= spk;
    out[(size_t)t * (BATCH * D_OUT) + b * D_OUT + o] = spk;
  }
}

extern "C" void kernel_launch(void* const* d_in, const int* in_sizes, int n_in,
                              void* d_out, int out_size, void* d_ws, size_t ws_size,
                              hipStream_t stream) {
  const float* x  = (const float*)d_in[0];
  const float* W1 = (const float*)d_in[1];
  const float* b1 = (const float*)d_in[2];
  const float* W2 = (const float*)d_in[3];
  const float* b2 = (const float*)d_in[4];
  float* out = (float*)d_out;

  char* ws = (char*)d_ws;
  size_t off = 0;
  _Float16* W1h = (_Float16*)(ws + off); off += (size_t)N1PAD * K1PAD * 2;      // 179,200
  _Float16* W2h = (_Float16*)(ws + off); off += (size_t)16 * K2PAD * 2;         //   4,096
  float*    CUR1 = (float*)(ws + off);   off += (size_t)M_TOTAL * C1STRIDE * 4;  // 45.9 MB
  _Float16* SPK1 = (_Float16*)(ws + off); off += (size_t)M_TOTAL * S1STRIDE * 2; // 26.2 MB
  float*    CUR2 = (float*)(ws + off);   off += (size_t)M_TOTAL * 16 * 4;        //  6.6 MB
  if (off > ws_size) return;  // insufficient scratch: deterministic no-op

  const int prep_n = N1PAD * K1PAD + 16 * K2PAD;
  snn_prep<<<(prep_n + 255) / 256, 256, 0, stream>>>(W1, W2, W1h, W2h);
  snn_gemm1<<<M_TOTAL / 128, 256, 0, stream>>>(x, W1h, CUR1);
  snn_lif1<<<(BATCH * D_H) / 256, 256, 0, stream>>>(CUR1, b1, SPK1);
  snn_gemm2<<<M_TOTAL / 128, 256, 0, stream>>>(SPK1, W2h, CUR2);
  snn_lif2<<<(BATCH * D_OUT) / 256, 256, 0, stream>>>(CUR2, b2, out);
}
